// MultiHeadAttention_19731079758556
// MI455X (gfx1250) — matrix operand order repaired and checked by reading, not compiled, not run on hardware
//
#include <hip/hip_runtime.h>
#include <hip/hip_bf16.h>
#include <stdint.h>

#define D_MODEL 1024
#define NQKV    3072
#define NHEADS  16
#define DHEAD   64
#define BATCH   2
#define SEQ     2048
#define NTOK    (BATCH * SEQ)

// GEMM tiling
#define TILE_M   64
#define TILE_N   128
#define KDEPTH   64
#define LDSTRIDE 72                       // padded row stride (elements)
#define LDS_A_ELE (TILE_M * LDSTRIDE)     // 4608
#define LDS_B_ELE (TILE_N * LDSTRIDE)     // 9216
#define LDS_BUF_ELE (LDS_A_ELE + LDS_B_ELE)

// softmax in exp2 domain: 1/sqrt(64) * log2(e)
#define SCALE_LOG2E 0.1803368801111204f

typedef __attribute__((ext_vector_type(16))) __bf16 v16bf;
typedef __attribute__((ext_vector_type(8)))  float  v8f;

#if defined(__AMDGCN__) && __has_builtin(__builtin_amdgcn_global_load_async_to_lds_b128) && __has_builtin(__builtin_amdgcn_s_wait_asynccnt)
#define HAVE_ASYNC_LDS 1
typedef int v4i_vs __attribute__((vector_size(16)));
typedef __attribute__((address_space(1))) v4i_vs* gptr4_t;
typedef __attribute__((address_space(3))) v4i_vs* lptr4_t;
#else
#define HAVE_ASYNC_LDS 0
#endif

__device__ __forceinline__ unsigned short f32_to_bf16(float f) {
  union { float f; uint32_t u; } v; v.f = f;
  uint32_t u = v.u;
  uint32_t r = u + 0x7FFFu + ((u >> 16) & 1u);      // round-to-nearest-even
  if ((u & 0x7F800000u) == 0x7F800000u) r = u;      // keep inf/nan payloads
  return (unsigned short)(r >> 16);
}

__device__ __forceinline__ v16bf frag_cast(uint4 lo, uint4 hi) {
  union { uint4 q[2]; v16bf v; } u;
  u.q[0] = lo; u.q[1] = hi;
  return u.v;
}

__device__ __forceinline__ v16bf load_frag(const unsigned short* p0,
                                           const unsigned short* p1) {
  return frag_cast(*(const uint4*)p0, *(const uint4*)p1);
}

__device__ __forceinline__ v8f wmma_bf16(v16bf a, v16bf b, v8f c) {
  return __builtin_amdgcn_wmma_f32_16x16x32_bf16(false, a, false, b, (short)0, c,
                                                 false, false);
}

// ---------------------------------------------------------------------------
// Fold LoRA into base weight, emit bf16 transposed weight: Wt[n][k]
// ---------------------------------------------------------------------------
__global__ void fold_lora_weights(const float* __restrict__ W,
                                  const float* __restrict__ A,
                                  const float* __restrict__ Bl,
                                  unsigned short* __restrict__ Wt,
                                  int K, int N) {
  int idx = blockIdx.x * blockDim.x + threadIdx.x;
  if (idx >= K * N) return;
  int n = idx / K;
  int k = idx - n * K;
  float acc = W[k * N + n];
#pragma unroll
  for (int r = 0; r < 8; ++r)
    acc += 2.0f * A[k * 8 + r] * Bl[r * N + n];   // LORA_SCALING = 2
  Wt[idx] = f32_to_bf16(acc);
}

__global__ void convert_f32_bf16(const float* __restrict__ src,
                                 unsigned short* __restrict__ dst, int count) {
  int i = blockIdx.x * blockDim.x + threadIdx.x;
  if (i < count) dst[i] = f32_to_bf16(src[i]);
}

// ---------------------------------------------------------------------------
// Cooperative tile fetch: A tile 64x64 bf16, B tile 128x64 bf16 (K-major rows)
// Async (ASYNCcnt); 6 async-load instructions per wave per tile.
// ---------------------------------------------------------------------------
__device__ __forceinline__ void issue_tile_loads(const unsigned short* Ag,
                                                 const unsigned short* Bg,
                                                 unsigned short* buf, int tid) {
#if HAVE_ASYNC_LDS
#pragma unroll
  for (int i = 0; i < 2; ++i) {                       // A: 64 rows * 8 chunks
    int t = tid + i * 256;
    int row = t >> 3, ch = t & 7;
    __builtin_amdgcn_global_load_async_to_lds_b128(
        (gptr4_t)(Ag + row * D_MODEL + ch * 8),
        (lptr4_t)(buf + row * LDSTRIDE + ch * 8), 0, 0);
  }
#pragma unroll
  for (int i = 0; i < 4; ++i) {                       // B: 128 rows * 8 chunks
    int t = tid + i * 256;
    int row = t >> 3, ch = t & 7;
    __builtin_amdgcn_global_load_async_to_lds_b128(
        (gptr4_t)(Bg + row * D_MODEL + ch * 8),
        (lptr4_t)(buf + LDS_A_ELE + row * LDSTRIDE + ch * 8), 0, 0);
  }
#else
#pragma unroll
  for (int i = 0; i < 2; ++i) {
    int t = tid + i * 256;
    int row = t >> 3, ch = t & 7;
    *(uint4*)(buf + row * LDSTRIDE + ch * 8) =
        *(const uint4*)(Ag + row * D_MODEL + ch * 8);
  }
#pragma unroll
  for (int i = 0; i < 4; ++i) {
    int t = tid + i * 256;
    int row = t >> 3, ch = t & 7;
    *(uint4*)(buf + LDS_A_ELE + row * LDSTRIDE + ch * 8) =
        *(const uint4*)(Bg + row * D_MODEL + ch * 8);
  }
#endif
}

// ---------------------------------------------------------------------------
// Double-buffered LDS-tiled GEMM core. Workgroup tile 64(M) x 128(N),
// 8 waves arranged 4(m) x 2(n); each wave accumulates a 16x64 tile (4 x v8f).
// All DS loads for a K-step are issued before the 4 back-to-back WMMAs.
// ---------------------------------------------------------------------------
__device__ __forceinline__ void gemm_core(const unsigned short* __restrict__ Ag,
                                          const unsigned short* __restrict__ Bg,
                                          unsigned short* lds, v8f acc[4]) {
  const int NKIT = D_MODEL / KDEPTH;               // 16
  int tid  = threadIdx.x;
  int lane = tid & 31;
  int w    = tid >> 5;
  int wm   = w & 3;
  int wn   = w >> 2;
  int l15  = lane & 15;
  int koA  = (lane & 16) ? 16 : 0;
  int koB  = (lane & 16) ? 16 : 0;

  issue_tile_loads(Ag, Bg, lds, tid);
  for (int it = 0; it < NKIT; ++it) {
    if (it + 1 < NKIT) {
      issue_tile_loads(Ag + (it + 1) * KDEPTH, Bg + (it + 1) * KDEPTH,
                       lds + ((it + 1) & 1) * LDS_BUF_ELE, tid);
#if HAVE_ASYNC_LDS
      __builtin_amdgcn_s_wait_asynccnt(6);         // current buffer landed
#endif
    } else {
#if HAVE_ASYNC_LDS
      __builtin_amdgcn_s_wait_asynccnt(0);
#endif
    }
    __syncthreads();                               // all waves' loads visible
    const unsigned short* bufA = lds + (it & 1) * LDS_BUF_ELE;
    const unsigned short* bufB = bufA + LDS_A_ELE;
    const unsigned short* arow = bufA + (wm * 16 + l15) * LDSTRIDE + koA;
    const unsigned short* brow = bufB + (wn * 64 + l15) * LDSTRIDE + koB;
#pragma unroll
    for (int ks = 0; ks < 2; ++ks) {
      int kk = ks * 32;
      v16bf a  = load_frag(arow + kk, arow + kk + 8);
      v16bf b0 = load_frag(brow + 0 * 16 * LDSTRIDE + kk,
                           brow + 0 * 16 * LDSTRIDE + kk + 8);
      v16bf b1 = load_frag(brow + 1 * 16 * LDSTRIDE + kk,
                           brow + 1 * 16 * LDSTRIDE + kk + 8);
      v16bf b2 = load_frag(brow + 2 * 16 * LDSTRIDE + kk,
                           brow + 2 * 16 * LDSTRIDE + kk + 8);
      v16bf b3 = load_frag(brow + 3 * 16 * LDSTRIDE + kk,
                           brow + 3 * 16 * LDSTRIDE + kk + 8);
      acc[0] = wmma_bf16(a, b0, acc[0]);
      acc[1] = wmma_bf16(a, b1, acc[1]);
      acc[2] = wmma_bf16(a, b2, acc[2]);
      acc[3] = wmma_bf16(a, b3, acc[3]);
    }
    __syncthreads();                               // protect buffer reuse
  }
}

// ---------------------------------------------------------------------------
// QKV GEMM: epilogue scatters into q[b,h,s,dh], k[b,h,s,dh], vT[b,h,dh,s]
// ---------------------------------------------------------------------------
__global__ void gemm_qkv(const unsigned short* __restrict__ xb,
                         const unsigned short* __restrict__ wt,
                         const float* __restrict__ bias,
                         unsigned short* __restrict__ qb,
                         unsigned short* __restrict__ kb,
                         unsigned short* __restrict__ vt) {
  __shared__ __align__(16) unsigned short lds[2 * LDS_BUF_ELE];
  const int NWG_N = NQKV / TILE_N;                 // 24
  int wgm = blockIdx.x / NWG_N;
  int wgn = blockIdx.x % NWG_N;

  v8f acc[4] = {{}, {}, {}, {}};
  gemm_core(xb + wgm * TILE_M * D_MODEL, wt + wgn * TILE_N * D_MODEL, lds, acc);

  int lane  = threadIdx.x & 31;
  int w     = threadIdx.x >> 5;
  int l15   = lane & 15;
  int half8 = (lane & 16) ? 8 : 0;
  int mt    = wgm * 4 + (w & 3);
  int nb    = wgn * TILE_N + (w >> 2) * 64;
  int p     = nb >> 10;                            // 0=q 1=k 2=v
  int h     = (nb & 1023) >> 6;
#pragma unroll
  for (int f = 0; f < 4; ++f) {
    int dh = f * 16 + l15;
    float bv = bias[nb + f * 16 + l15];
#pragma unroll
    for (int r = 0; r < 8; ++r) {
      int token = mt * 16 + r + half8;
      int bb = token >> 11;
      int s  = token & (SEQ - 1);
      unsigned short val = f32_to_bf16(acc[f][r] + bv);
      if (p == 0)      qb[((bb * NHEADS + h) * SEQ + s) * DHEAD + dh] = val;
      else if (p == 1) kb[((bb * NHEADS + h) * SEQ + s) * DHEAD + dh] = val;
      else             vt[((bb * NHEADS + h) * DHEAD + dh) * SEQ + s] = val;
    }
  }
}

// ---------------------------------------------------------------------------
// Flash attention: one wave per 16-row Q tile, online softmax (exp2 domain),
// key blocks of 32.  V fragments are fetched before the softmax so their
// global latency hides under the VALU work and the P LDS round-trip.
// ---------------------------------------------------------------------------
__global__ void attention(const unsigned short* __restrict__ qbuf,
                          const unsigned short* __restrict__ kbuf,
                          const unsigned short* __restrict__ vtbuf,
                          const int* __restrict__ amask,
                          unsigned short* __restrict__ ctx) {
  __shared__ __align__(16) unsigned short plds[4][16 * 32];
  int w    = threadIdx.x >> 5;
  int lane = threadIdx.x & 31;
  int wid  = blockIdx.x * 4 + w;
  int bh   = wid >> 7;
  int qt   = wid & 127;
  int b    = bh >> 4, h = bh & 15;
  int qs   = qt * 16;

  const unsigned short* qh = qbuf + (long)bh * SEQ * DHEAD;
  const unsigned short* kh = kbuf + (long)bh * SEQ * DHEAD;
  const unsigned short* vh = vtbuf + (long)bh * DHEAD * SEQ;
  const int* mrow = amask + b * SEQ;
  unsigned short* pl = &plds[w][0];

  int l15   = lane & 15;
  int koA   = (lane & 16) ? 16 : 0;
  int koB   = (lane & 16) ? 16 : 0;
  int half8 = (lane & 16) ? 8 : 0;

  const unsigned short* qrow = qh + (qs + l15) * DHEAD;
  v16bf qa0 = load_frag(qrow + koA, qrow + 8 + koA);
  v16bf qa1 = load_frag(qrow + 32 + koA, qrow + 40 + koA);

  float mrun[8], lrun[8];
  v8f o[4] = {{}, {}, {}, {}};
#pragma unroll
  for (int r = 0; r < 8; ++r) { mrun[r] = -1.0e30f; lrun[r] = 0.0f; }

  int nblk = (qs + 16 + 31) >> 5;
  for (int blk = 0; blk < nblk; ++blk) {
    int k0 = blk * 32;
    float s0[8], s1[8];
#pragma unroll
    for (int sub = 0; sub < 2; ++sub) {
      int kbase = k0 + sub * 16;
      const unsigned short* krow = kh + (kbase + l15) * DHEAD + koB;
      v16bf kb0 = load_frag(krow, krow + 8);
      v16bf kb1 = load_frag(krow + 32, krow + 40);
      v8f sc = {};
      sc = wmma_bf16(qa0, kb0, sc);
      sc = wmma_bf16(qa1, kb1, sc);
      int nkey = kbase + l15;
      int pad  = mrow[nkey];
      float* sout = sub ? s1 : s0;
#pragma unroll
      for (int r = 0; r < 8; ++r) {
        float v = sc[r];
        int qm = qs + r + half8;
        if (nkey > qm || pad == 0) v = -1.0e9f;    // mask BEFORE scaling (ref)
        sout[r] = v * SCALE_LOG2E;                 // into exp2 domain
      }
    }
    // fetch V fragments early: independent of softmax below
    const unsigned short* vrow = vh + l15 * SEQ + k0 + koB;
    v16bf vb0 = load_frag(vrow + 0 * 16 * SEQ, vrow + 0 * 16 * SEQ + 8);
    v16bf vb1 = load_frag(vrow + 1 * 16 * SEQ, vrow + 1 * 16 * SEQ + 8);
    v16bf vb2 = load_frag(vrow + 2 * 16 * SEQ, vrow + 2 * 16 * SEQ + 8);
    v16bf vb3 = load_frag(vrow + 3 * 16 * SEQ, vrow + 3 * 16 * SEQ + 8);
#pragma unroll
    for (int r = 0; r < 8; ++r) {
      float bm = fmaxf(s0[r], s1[r]);
      bm = fmaxf(bm, __shfl_xor(bm, 1, 32));
      bm = fmaxf(bm, __shfl_xor(bm, 2, 32));
      bm = fmaxf(bm, __shfl_xor(bm, 4, 32));
      bm = fmaxf(bm, __shfl_xor(bm, 8, 32));
      float mn   = fmaxf(mrun[r], bm);
      float corr = exp2f(mrun[r] - mn);
      mrun[r] = mn;
      float p0 = exp2f(s0[r] - mn);
      float p1 = exp2f(s1[r] - mn);
      float rs = p0 + p1;
      rs += __shfl_xor(rs, 1, 32);
      rs += __shfl_xor(rs, 2, 32);
      rs += __shfl_xor(rs, 4, 32);
      rs += __shfl_xor(rs, 8, 32);
      lrun[r] = lrun[r] * corr + rs;
#pragma unroll
      for (int f = 0; f < 4; ++f) o[f][r] *= corr;
      int m = r + half8;
      pl[m * 32 + l15]      = f32_to_bf16(p0);
      pl[m * 32 + 16 + l15] = f32_to_bf16(p1);
    }
    const unsigned short* prow = pl + l15 * 32;
    v16bf pa = load_frag(prow + koA, prow + 8 + koA);
    o[0] = wmma_bf16(pa, vb0, o[0]);
    o[1] = wmma_bf16(pa, vb1, o[1]);
    o[2] = wmma_bf16(pa, vb2, o[2]);
    o[3] = wmma_bf16(pa, vb3, o[3]);
  }

#pragma unroll
  for (int r = 0; r < 8; ++r) {
    float inv = 1.0f / lrun[r];
    int token = b * SEQ + qs + r + half8;
#pragma unroll
    for (int f = 0; f < 4; ++f) {
      ctx[(long)token * D_MODEL + h * DHEAD + f * 16 + l15] =
          f32_to_bf16(o[f][r] * inv);
    }
  }
}

// ---------------------------------------------------------------------------
// Output projection: fp32 result
// ---------------------------------------------------------------------------
__global__ void gemm_proj(const unsigned short* __restrict__ cb,
                          const unsigned short* __restrict__ wt,
                          const float* __restrict__ bias,
                          float* __restrict__ out) {
  __shared__ __align__(16) unsigned short lds[2 * LDS_BUF_ELE];
  const int NWG_N = D_MODEL / TILE_N;              // 8
  int wgm = blockIdx.x / NWG_N;
  int wgn = blockIdx.x % NWG_N;

  v8f acc[4] = {{}, {}, {}, {}};
  gemm_core(cb + wgm * TILE_M * D_MODEL, wt + wgn * TILE_N * D_MODEL, lds, acc);

  int lane  = threadIdx.x & 31;
  int w     = threadIdx.x >> 5;
  int l15   = lane & 15;
  int half8 = (lane & 16) ? 8 : 0;
  int mt    = wgm * 4 + (w & 3);
  int nb    = wgn * TILE_N + (w >> 2) * 64;
#pragma unroll
  for (int f = 0; f < 4; ++f) {
    int col = nb + f * 16 + l15;
    float bv = bias[col];
#pragma unroll
    for (int r = 0; r < 8; ++r) {
      int token = mt * 16 + r + half8;
      out[(long)token * D_MODEL + col] = acc[f][r] + bv;
    }
  }
}

// ---------------------------------------------------------------------------
extern "C" void kernel_launch(void* const* d_in, const int* in_sizes, int n_in,
                              void* d_out, int out_size, void* d_ws, size_t ws_size,
                              hipStream_t stream) {
  const float* x     = (const float*)d_in[0];
  const int*   amask = (const int*)d_in[1];
  const float* Wqkv  = (const float*)d_in[2];
  const float* bqkv  = (const float*)d_in[3];
  const float* Aq    = (const float*)d_in[4];
  const float* Bq    = (const float*)d_in[5];
  const float* Wp    = (const float*)d_in[6];
  const float* bp    = (const float*)d_in[7];
  const float* Ap    = (const float*)d_in[8];
  const float* Bp    = (const float*)d_in[9];
  float* out = (float*)d_out;

  char* ws = (char*)d_ws;
  unsigned short* xb    = (unsigned short*)ws; ws += (size_t)NTOK * D_MODEL * 2;
  unsigned short* wqkvt = (unsigned short*)ws; ws += (size_t)NQKV * D_MODEL * 2;
  unsigned short* wpt   = (unsigned short*)ws; ws += (size_t)D_MODEL * D_MODEL * 2;
  unsigned short* qb    = (unsigned short*)ws; ws += (size_t)NTOK * D_MODEL * 2;
  unsigned short* kbuf  = (unsigned short*)ws; ws += (size_t)NTOK * D_MODEL * 2;
  unsigned short* vt    = (unsigned short*)ws; ws += (size_t)NTOK * D_MODEL * 2;
  unsigned short* ctx   = (unsigned short*)ws; ws += (size_t)NTOK * D_MODEL * 2;

  fold_lora_weights<<<(D_MODEL * NQKV) / 256, 256, 0, stream>>>(
      Wqkv, Aq, Bq, wqkvt, D_MODEL, NQKV);
  fold_lora_weights<<<(D_MODEL * D_MODEL) / 256, 256, 0, stream>>>(
      Wp, Ap, Bp, wpt, D_MODEL, D_MODEL);
  convert_f32_bf16<<<(NTOK * D_MODEL) / 256, 256, 0, stream>>>(
      x, xb, NTOK * D_MODEL);

  // 64 m-wgs * 24 n-wgs
  gemm_qkv<<<64 * (NQKV / TILE_N), 256, 0, stream>>>(xb, wqkvt, bqkv, qb, kbuf, vt);
  // 2*16 heads * 128 q-tiles = 4096 waves, 4 waves/block
  attention<<<1024, 128, 0, stream>>>(qb, kbuf, vt, amask, ctx);
  // 64 m-wgs * 8 n-wgs
  gemm_proj<<<64 * (D_MODEL / TILE_N), 256, 0, stream>>>(ctx, wpt, bp, out);
}
